// ResourceEncoder_11613591568819
// MI455X (gfx1250) — matrix operand order repaired and checked by reading, not compiled, not run on hardware
//
#include <hip/hip_runtime.h>

typedef __bf16 bf16_t;
typedef __attribute__((ext_vector_type(16))) __bf16 v16bf;
typedef __attribute__((ext_vector_type(8)))  __bf16 v8bf;
typedef __attribute__((ext_vector_type(8)))  float  v8f;
typedef __attribute__((ext_vector_type(4)))  float  v4f;
typedef __attribute__((ext_vector_type(4)))  unsigned int v4u;

#define DIM 128
#define NT  8   // 128 output cols / 16

static __device__ __forceinline__ void fatomic_add(float* p, float v) {
    __hip_atomic_fetch_add(p, v, __ATOMIC_RELAXED, __HIP_MEMORY_SCOPE_AGENT);
}

// ---------------- utility kernels ----------------

__global__ void zero_f32_kernel(float* __restrict__ p, long long n) {
    long long t = (long long)blockIdx.x * blockDim.x + threadIdx.x;
    if (t < n) p[t] = 0.0f;
}

__global__ void cvt_bf16_kernel(const float* __restrict__ in, bf16_t* __restrict__ out, long long n) {
    long long t = (long long)blockIdx.x * blockDim.x + threadIdx.x;
    if (t < n) out[t] = (bf16_t)in[t];
}

__global__ void count_deg_kernel(const long long* __restrict__ dst, float* __restrict__ cnt, int E) {
    int t = blockIdx.x * blockDim.x + threadIdx.x;
    if (t < E) fatomic_add(&cnt[dst[t]], 1.0f);
}

// Pack a row-major f32 weight matrix [Kreal x 128] into WMMA B-fragment order,
// each lane's 32B fragment split into two 16B chunks stored SoA (stride-16B,
// conflict-minimal ds_load_b128):
//   pk[(((kb*NT + nt)*2 + ch)*32 + lane)*8 + j] =
//       bf16(W[kb*32 + 8*(lane>>4) + ch*16 + j][nt*16 + (lane&15)])
__global__ void pack_frag_kernel(const float* __restrict__ W, bf16_t* __restrict__ pk,
                                 int Kreal, int KB) {
    int t = blockIdx.x * blockDim.x + threadIdx.x;
    int total = KB << 12;                 // KB * 8 * 2 * 32 * 8
    if (t >= total) return;
    int j    =  t        & 7;
    int lane = (t >> 3)  & 31;
    int ch   = (t >> 8)  & 1;
    int nt   = (t >> 9)  & 7;
    int kb   =  t >> 12;
    int K    = kb * 32 + 8 * (lane >> 4) + ch * 16 + j;
    int col  = nt * 16 + (lane & 15);
    float v  = (K < Kreal) ? W[(long long)K * DIM + col] : 0.0f;
    pk[t] = (bf16_t)v;
}

// ---------------- fragment loaders ----------------

// B fragment (v16bf) from LDS-staged packed weights
static __device__ __forceinline__ v16bf ld_bfrag(const bf16_t* sW, int frag, int lane) {
    const v8bf* sv = reinterpret_cast<const v8bf*>(sW);
    v8bf b0 = sv[frag * 64 + lane];
    v8bf b1 = sv[frag * 64 + 32 + lane];
    return __builtin_shufflevector(b0, b1, 0,1,2,3,4,5,6,7,8,9,10,11,12,13,14,15);
}

// A fragment from a bf16 feature row (K block kb of 32)
static __device__ __forceinline__ v16bf ld_afrag_bf16(const bf16_t* row, int kb, int half) {
    v8bf a0 = *reinterpret_cast<const v8bf*>(row + kb * 32 + 8 * half);
    v8bf a1 = *reinterpret_cast<const v8bf*>(row + kb * 32 + 16 + 8 * half);
    return __builtin_shufflevector(a0, a1, 0,1,2,3,4,5,6,7,8,9,10,11,12,13,14,15);
}

// A fragment from the 16-wide f32 edge-attr row (upper 16 K padded with zeros)
static __device__ __forceinline__ v16bf ld_afrag_attr(const float* arow, int half) {
    v4f f0 = *reinterpret_cast<const v4f*>(arow + 8 * half);
    v4f f1 = *reinterpret_cast<const v4f*>(arow + 8 * half + 4);
    v16bf a;
#pragma unroll
    for (int j = 0; j < 4; ++j) {
        a[j]      = (bf16_t)f0[j];
        a[4 + j]  = (bf16_t)f1[j];
        a[8 + j]  = (bf16_t)0.0f;
        a[12 + j] = (bf16_t)0.0f;
    }
    return a;
}

// A fragment from the f32 aggregate row, scaled by 1/deg (K block kb4 of 32)
static __device__ __forceinline__ v16bf ld_afrag_agg(const float* arow, int kb4, int half, float inv) {
    int o0 = kb4 * 32 + 8 * half;
    int o1 = kb4 * 32 + 16 + 8 * half;
    v4f f0 = *reinterpret_cast<const v4f*>(arow + o0);
    v4f f1 = *reinterpret_cast<const v4f*>(arow + o0 + 4);
    v4f f2 = *reinterpret_cast<const v4f*>(arow + o1);
    v4f f3 = *reinterpret_cast<const v4f*>(arow + o1 + 4);
    v16bf a;
#pragma unroll
    for (int j = 0; j < 4; ++j) {
        a[j]      = (bf16_t)(f0[j] * inv);
        a[4 + j]  = (bf16_t)(f1[j] * inv);
        a[8 + j]  = (bf16_t)(f2[j] * inv);
        a[12 + j] = (bf16_t)(f3[j] * inv);
    }
    return a;
}

// ---------------- edge message + scatter kernel ----------------
// One wave handles 16 edges: m = relu([x[src], edge_attr] @ W + b) -> atomic add into agg[dst].
// K padded to 160. Software-pipelined: B frag f+1 loaded before WMMA f, A frag one kb ahead.
__global__ void __launch_bounds__(256) edge_msg_kernel(
    const bf16_t* __restrict__ xin, const float* __restrict__ eattr,
    const long long* __restrict__ sidx, const long long* __restrict__ didx,
    const bf16_t* __restrict__ Wpk, const float* __restrict__ bias,
    float* __restrict__ agg, int ntiles)
{
    __shared__ __align__(16) bf16_t sW[5 * NT * 2 * 32 * 8];   // 40KB

    {
        const v4u* g = reinterpret_cast<const v4u*>(Wpk);
        v4u*       s = reinterpret_cast<v4u*>(sW);
#pragma unroll
        for (int i = 0; i < 10; ++i)
            s[i * 256 + threadIdx.x] = g[i * 256 + threadIdx.x];
    }
    __syncthreads();

    int wid = blockIdx.x * (blockDim.x >> 5) + (threadIdx.x >> 5);
    if (wid >= ntiles) return;
    int lane = threadIdx.x & 31;
    int m    = lane & 15;
    int half = lane >> 4;

    long long e   = (long long)wid * 16 + m;
    long long src = sidx[e];
    const bf16_t* xrow = xin + src * DIM;
    const float*  arow = eattr + e * 16;

    // prefetch scatter indices + bias early; latency hides behind the GEMM
    long long drow[8];
#pragma unroll
    for (int r = 0; r < 8; ++r) drow[r] = didx[(long long)wid * 16 + 8 * half + r];
    float bv[NT];
#pragma unroll
    for (int nt = 0; nt < NT; ++nt) bv[nt] = bias[nt * 16 + m];

    v8f c[NT];
    v8f z8 = {0.f,0.f,0.f,0.f,0.f,0.f,0.f,0.f};
#pragma unroll
    for (int nt = 0; nt < NT; ++nt) c[nt] = z8;

    v16bf a_cur = ld_afrag_bf16(xrow, 0, half);
    v16bf b_cur = ld_bfrag(sW, 0, lane);

#pragma unroll
    for (int kb = 0; kb < 5; ++kb) {
        v16bf a_nxt;
        if (kb < 3)       a_nxt = ld_afrag_bf16(xrow, kb + 1, half);
        else if (kb == 3) a_nxt = ld_afrag_attr(arow, half);
#pragma unroll
        for (int nt = 0; nt < NT; ++nt) {
            int f = kb * NT + nt;
            v16bf b_nxt = (f < 5 * NT - 1) ? ld_bfrag(sW, f + 1, lane) : b_cur;
            c[nt] = __builtin_amdgcn_wmma_f32_16x16x32_bf16(false, a_cur, false, b_cur, (short)0,
                                                            c[nt], false, false);
            b_cur = b_nxt;
        }
        if (kb < 4) a_cur = a_nxt;
    }

    // epilogue: bias + relu + scatter (C layout: VGPR r -> row r + 8*half, col nt*16 + m)
#pragma unroll
    for (int nt = 0; nt < NT; ++nt) {
#pragma unroll
        for (int r = 0; r < 8; ++r) {
            float v = c[nt][r] + bv[nt];
            v = v > 0.0f ? v : 0.0f;
            fatomic_add(&agg[drow[r] * DIM + nt * 16 + m], v);
        }
    }
}

// ---------------- node update kernel ----------------
// One wave handles 16 nodes: out = relu([h, agg/max(cnt,1)] @ U). K = 256 -> 8 K-steps.
__global__ void __launch_bounds__(256) node_upd_kernel(
    const bf16_t* __restrict__ hin, const float* __restrict__ agg,
    const float* __restrict__ cnt, const bf16_t* __restrict__ Upk,
    bf16_t* __restrict__ hout, float* __restrict__ fout, int ntiles, int last)
{
    __shared__ __align__(16) bf16_t sU[8 * NT * 2 * 32 * 8];   // 64KB

    {
        const v4u* g = reinterpret_cast<const v4u*>(Upk);
        v4u*       s = reinterpret_cast<v4u*>(sU);
#pragma unroll
        for (int i = 0; i < 16; ++i)
            s[i * 256 + threadIdx.x] = g[i * 256 + threadIdx.x];
    }
    __syncthreads();

    int wid = blockIdx.x * (blockDim.x >> 5) + (threadIdx.x >> 5);
    if (wid >= ntiles) return;
    int lane = threadIdx.x & 31;
    int m    = lane & 15;
    int half = lane >> 4;

    long long nrow = (long long)wid * 16 + m;
    float cd  = cnt[nrow];
    float inv = 1.0f / (cd > 1.0f ? cd : 1.0f);
    const bf16_t* hrow = hin + nrow * DIM;
    const float*  arow = agg + nrow * DIM;

    v8f c[NT];
    v8f z8 = {0.f,0.f,0.f,0.f,0.f,0.f,0.f,0.f};
#pragma unroll
    for (int nt = 0; nt < NT; ++nt) c[nt] = z8;

    v16bf a_cur = ld_afrag_bf16(hrow, 0, half);
    v16bf b_cur = ld_bfrag(sU, 0, lane);

#pragma unroll
    for (int kb = 0; kb < 8; ++kb) {
        v16bf a_nxt;
        if (kb < 3)      a_nxt = ld_afrag_bf16(hrow, kb + 1, half);
        else if (kb < 7) a_nxt = ld_afrag_agg(arow, kb + 1 - 4, half, inv);
#pragma unroll
        for (int nt = 0; nt < NT; ++nt) {
            int f = kb * NT + nt;
            v16bf b_nxt = (f < 8 * NT - 1) ? ld_bfrag(sU, f + 1, lane) : b_cur;
            c[nt] = __builtin_amdgcn_wmma_f32_16x16x32_bf16(false, a_cur, false, b_cur, (short)0,
                                                            c[nt], false, false);
            b_cur = b_nxt;
        }
        if (kb < 7) a_cur = a_nxt;
    }

    // relu + store (bf16 for next layer; f32 to d_out on the final layer)
#pragma unroll
    for (int nt = 0; nt < NT; ++nt) {
#pragma unroll
        for (int r = 0; r < 8; ++r) {
            float v = c[nt][r] > 0.0f ? c[nt][r] : 0.0f;
            long long n = (long long)wid * 16 + 8 * half + r;
            long long idx = n * DIM + nt * 16 + m;
            hout[idx] = (bf16_t)v;
            if (last) fout[idx] = v;
        }
    }
}

// ---------------- host launcher ----------------
extern "C" void kernel_launch(void* const* d_in, const int* in_sizes, int n_in,
                              void* d_out, int out_size, void* d_ws, size_t ws_size,
                              hipStream_t stream) {
    const float*     x     = (const float*)d_in[0];
    const long long* eidx  = (const long long*)d_in[1];   // int64 [2, E]
    const float*     eattr = (const float*)d_in[2];
    const float*     W1    = (const float*)d_in[3];
    const float*     b1    = (const float*)d_in[4];
    const float*     U1    = (const float*)d_in[5];
    const float*     W2    = (const float*)d_in[6];
    const float*     b2    = (const float*)d_in[7];
    const float*     U2    = (const float*)d_in[8];

    const int N = in_sizes[0] / DIM;   // 50000
    const int E = in_sizes[1] / 2;     // 600000
    const long long* src = eidx;
    const long long* dst = eidx + E;

    size_t off = 0;
    auto alloc = [&](size_t bytes) -> void* {
        void* p = (char*)d_ws + off;
        off = (off + bytes + 255) & ~(size_t)255;
        return p;
    };
    bf16_t* xb   = (bf16_t*)alloc((size_t)N * DIM * 2);
    bf16_t* hb   = (bf16_t*)alloc((size_t)N * DIM * 2);
    float*  agg  = (float*) alloc((size_t)N * DIM * 4);
    float*  cnt  = (float*) alloc((size_t)N * 4);
    bf16_t* Wpk1 = (bf16_t*)alloc((size_t)5 * 4096 * 2);
    bf16_t* Wpk2 = (bf16_t*)alloc((size_t)5 * 4096 * 2);
    bf16_t* Upk1 = (bf16_t*)alloc((size_t)8 * 4096 * 2);
    bf16_t* Upk2 = (bf16_t*)alloc((size_t)8 * 4096 * 2);
    (void)ws_size; (void)n_in; (void)out_size;

    long long nx = (long long)N * DIM;
    int thr = 256;

    cvt_bf16_kernel<<<(int)((nx + thr - 1) / thr), thr, 0, stream>>>(x, xb, nx);
    pack_frag_kernel<<<(5 * 4096 + thr - 1) / thr, thr, 0, stream>>>(W1, Wpk1, 144, 5);
    pack_frag_kernel<<<(5 * 4096 + thr - 1) / thr, thr, 0, stream>>>(W2, Wpk2, 144, 5);
    pack_frag_kernel<<<(8 * 4096 + thr - 1) / thr, thr, 0, stream>>>(U1, Upk1, 256, 8);
    pack_frag_kernel<<<(8 * 4096 + thr - 1) / thr, thr, 0, stream>>>(U2, Upk2, 256, 8);

    zero_f32_kernel<<<(N + thr - 1) / thr, thr, 0, stream>>>(cnt, N);
    count_deg_kernel<<<(E + thr - 1) / thr, thr, 0, stream>>>(dst, cnt, E);

    int etiles = E / 16, ntile = N / 16;
    int eblk = (etiles + 7) / 8, nblk = (ntile + 7) / 8;   // 8 waves / block
    float* fout = (float*)d_out;

    // layer 1 (W1/b1/U1): xb -> hb
    zero_f32_kernel<<<(int)((nx + thr - 1) / thr), thr, 0, stream>>>(agg, nx);
    edge_msg_kernel<<<eblk, thr, 0, stream>>>(xb, eattr, src, dst, Wpk1, b1, agg, etiles);
    node_upd_kernel<<<nblk, thr, 0, stream>>>(xb, agg, cnt, Upk1, hb, fout, ntile, 0);

    // layer 2 (W2/b2/U2): hb -> xb
    zero_f32_kernel<<<(int)((nx + thr - 1) / thr), thr, 0, stream>>>(agg, nx);
    edge_msg_kernel<<<eblk, thr, 0, stream>>>(hb, eattr, src, dst, Wpk2, b2, agg, etiles);
    node_upd_kernel<<<nblk, thr, 0, stream>>>(hb, agg, cnt, Upk2, xb, fout, ntile, 0);

    // layer 3 (W2/b2/U2): xb -> hb + f32 d_out
    zero_f32_kernel<<<(int)((nx + thr - 1) / thr), thr, 0, stream>>>(agg, nx);
    edge_msg_kernel<<<eblk, thr, 0, stream>>>(xb, eattr, src, dst, Wpk2, b2, agg, etiles);
    node_upd_kernel<<<nblk, thr, 0, stream>>>(xb, agg, cnt, Upk2, hb, fout, ntile, 1);
}
